// lstm_seq2seq_2010044695030
// MI455X (gfx1250) — matrix operand order repaired and checked by reading, not compiled, not run on hardware
//
#include <hip/hip_runtime.h>
#include <hip/hip_bf16.h>

// ---------------------------------------------------------------------------
// Problem constants (match reference)
// ---------------------------------------------------------------------------
constexpr int I_SIZE = 512;
constexpr int Hh     = 1024;       // hidden size H
constexpr int HHs    = 512;        // H/2
constexpr int Pp     = 16;
constexpr int SEQL   = 256;
constexpr int BATCH  = 128;
constexpr int TGTL   = 128;
constexpr int DIN    = I_SIZE + HHs;   // 1024
constexpr int G4H    = 4 * Hh;         // 4096

typedef __attribute__((ext_vector_type(16))) __bf16 v16bf;
typedef __attribute__((ext_vector_type(8)))  __bf16 v8bf;
typedef __attribute__((ext_vector_type(8)))  float  v8f;

__device__ __forceinline__ __bf16 f2bf(float f) {
  union { float f; unsigned u; } x; x.f = f;
  unsigned r = x.u + 0x7FFFu + ((x.u >> 16) & 1u);   // round-to-nearest-even
  union { unsigned short s; __bf16 b; } y;
  y.s = (unsigned short)(r >> 16);
  return y.b;
}

__device__ __forceinline__ float sigm(float x) { return 1.0f / (1.0f + __expf(-x)); }

// ---------------------------------------------------------------------------
// WMMA GEMM:  C(M=128 x N) = A0(128 x K0) * W0^T + [A1(128 x K1) * W1^T] + bias
// W is row-major (N x K) => row n of W is column n of the B-matrix.
// Register blocking: each wave computes a 32x64 tile = 2(M) x 4(N) subtiles
// of 16x16, so 2 A-frags + 4 B-frags feed 8 v_wmma_f32_16x16x32_bf16 per
// K-step (1.5 16B-loads per WMMA).  Block = 4 waves -> 32x256 tile.
// Fragment layouts per CDNA5 ISA 7.12.2 (wave32).
// ---------------------------------------------------------------------------
__device__ __forceinline__ v16bf load_a_frag(const __bf16* __restrict__ arow,
                                             int k, int hf) {
  union { v16bf v; v8bf h[2]; } a;
  a.h[0] = *(const v8bf*)(arow + k + hf * 8);        // K {0..7} | {8..15}
  a.h[1] = *(const v8bf*)(arow + k + 16 + hf * 8);   // K {16..23} | {24..31}
  return a.v;
}
__device__ __forceinline__ v16bf load_b_frag(const __bf16* __restrict__ wrow,
                                             int k, int hf) {
  union { v16bf v; v8bf h[2]; } b;
  const __bf16* wp = wrow + k + hf * 8;              // K {0..7,16..23} | {8..15,24..31}
  b.h[0] = *(const v8bf*)(wp);
  b.h[1] = *(const v8bf*)(wp + 16);
  return b.v;
}

__global__ __launch_bounds__(128) void gemm_wmma_kernel(
    const __bf16* __restrict__ A0, int lda0, int K0, const __bf16* __restrict__ W0,
    const __bf16* __restrict__ A1, int lda1, int K1, const __bf16* __restrict__ W1,
    const float* __restrict__ bias, float* __restrict__ C, int ldc) {
  const int lane = threadIdx.x & 31;
  const int wave = threadIdx.x >> 5;
  const int hf   = lane >> 4;       // lane half (0: lanes 0-15, 1: lanes 16-31)
  const int l15  = lane & 15;
  const int m0   = blockIdx.y * 32;                 // 2 M-subtiles: m0, m0+16
  const int n0   = blockIdx.x * 256 + wave * 64;    // 4 N-subtiles

  v8f acc[2][4];
#pragma unroll
  for (int j = 0; j < 4; ++j) {
    const float bv = bias ? bias[n0 + j * 16 + l15] : 0.0f;
#pragma unroll
    for (int mi = 0; mi < 2; ++mi)
#pragma unroll
      for (int i = 0; i < 8; ++i) acc[mi][j][i] = bv;
  }

  const __bf16* ar0 = A0 + (size_t)(m0 + l15) * lda0;
  const __bf16* ar1 = A0 + (size_t)(m0 + 16 + l15) * lda0;
  const __bf16* wr[4];
#pragma unroll
  for (int j = 0; j < 4; ++j) wr[j] = W0 + (size_t)(n0 + j * 16 + l15) * K0;

  for (int k = 0; k < K0; k += 32) {
    v16bf a0 = load_a_frag(ar0, k, hf);
    v16bf a1 = load_a_frag(ar1, k, hf);
#pragma unroll
    for (int j = 0; j < 4; ++j) {
      v16bf b = load_b_frag(wr[j], k, hf);
      acc[0][j] = __builtin_amdgcn_wmma_f32_16x16x32_bf16(false, a0, false, b,
                                                          (short)0, acc[0][j], false, false);
      acc[1][j] = __builtin_amdgcn_wmma_f32_16x16x32_bf16(false, a1, false, b,
                                                          (short)0, acc[1][j], false, false);
    }
  }

  if (W1) {
    const __bf16* br0 = A1 + (size_t)(m0 + l15) * lda1;
    const __bf16* br1 = A1 + (size_t)(m0 + 16 + l15) * lda1;
    const __bf16* wr1[4];
#pragma unroll
    for (int j = 0; j < 4; ++j) wr1[j] = W1 + (size_t)(n0 + j * 16 + l15) * K1;
    for (int k = 0; k < K1; k += 32) {
      v16bf a0 = load_a_frag(br0, k, hf);
      v16bf a1 = load_a_frag(br1, k, hf);
#pragma unroll
      for (int j = 0; j < 4; ++j) {
        v16bf b = load_b_frag(wr1[j], k, hf);
        acc[0][j] = __builtin_amdgcn_wmma_f32_16x16x32_bf16(false, a0, false, b,
                                                            (short)0, acc[0][j], false, false);
        acc[1][j] = __builtin_amdgcn_wmma_f32_16x16x32_bf16(false, a1, false, b,
                                                            (short)0, acc[1][j], false, false);
      }
    }
  }

  // C/D layout: VGPR i -> row (subtile_m + hf*8 + i), col (subtile_n + l15)
#pragma unroll
  for (int mi = 0; mi < 2; ++mi) {
#pragma unroll
    for (int i = 0; i < 8; ++i) {
      float* crow = C + (size_t)(m0 + mi * 16 + hf * 8 + i) * ldc + l15;
#pragma unroll
      for (int j = 0; j < 4; ++j) crow[n0 + j * 16] = acc[mi][j][i];
    }
  }
}

// ---------------------------------------------------------------------------
// LSTM cell elementwise: gates(B x 4H, order i,f,g,o) + c  ->  c(fp32), h(bf16)
// ---------------------------------------------------------------------------
__global__ void lstm_cell_kernel(const float* __restrict__ gates,
                                 float* __restrict__ c,
                                 __bf16* __restrict__ h) {
  int idx = blockIdx.x * blockDim.x + threadIdx.x;
  if (idx >= BATCH * Hh) return;
  int b = idx / Hh, j = idx - b * Hh;
  const float* g = gates + (size_t)b * G4H;
  float gi = g[j], gf = g[Hh + j], gg = g[2 * Hh + j], go = g[3 * Hh + j];
  float cn = sigm(gf) * c[idx] + sigm(gi) * tanhf(gg);
  c[idx] = cn;
  h[idx] = f2bf(sigm(go) * tanhf(cn));
}

// ---------------------------------------------------------------------------
// Small helper kernels
// ---------------------------------------------------------------------------
__global__ void bf16_cast_kernel(const float* __restrict__ s, __bf16* __restrict__ d, int n) {
  int i = blockIdx.x * blockDim.x + threadIdx.x;
  if (i < n) d[i] = f2bf(s[i]);
}

__global__ void bf16_cast_strided_kernel(const float* __restrict__ s, int slds,
                                         __bf16* __restrict__ d, int dlds,
                                         int rows, int cols) {
  int i = blockIdx.x * blockDim.x + threadIdx.x;
  if (i >= rows * cols) return;
  int r = i / cols, cc = i - r * cols;
  d[(size_t)r * dlds + cc] = f2bf(s[(size_t)r * slds + cc]);
}

__global__ void vec_add_kernel(const float* __restrict__ a, const float* __restrict__ b,
                               float* __restrict__ o, int n) {
  int i = blockIdx.x * blockDim.x + threadIdx.x;
  if (i < n) o[i] = a[i] + b[i];
}

__global__ void fill_f32_kernel(float* p, int n) {
  int i = blockIdx.x * blockDim.x + threadIdx.x;
  if (i < n) p[i] = 0.0f;
}

__global__ void fill_bf16_kernel(__bf16* p, int n) {
  int i = blockIdx.x * blockDim.x + threadIdx.x;
  if (i < n) p[i] = f2bf(0.0f);
}

// par_dense layer 1: z = LeakyReLU(pars_last @ w1^T + b1), K=16, out bf16 (B x HH)
__global__ void pd1_kernel(const float* __restrict__ pars_last,  // B x P
                           const float* __restrict__ w1,         // HH x P
                           const float* __restrict__ b1,
                           __bf16* __restrict__ z) {
  int i = blockIdx.x * blockDim.x + threadIdx.x;
  if (i >= BATCH * HHs) return;
  int b = i / HHs, n = i - b * HHs;
  float s = b1[n];
  const float* pv = pars_last + (size_t)b * Pp;
  const float* wv = w1 + (size_t)n * Pp;
#pragma unroll
  for (int k = 0; k < Pp; ++k) s += pv[k] * wv[k];
  s = s > 0.0f ? s : 0.01f * s;
  z[i] = f2bf(s);
}

// ---------------------------------------------------------------------------
// Host orchestration
// ---------------------------------------------------------------------------
extern "C" void kernel_launch(void* const* d_in, const int* in_sizes, int n_in,
                              void* d_out, int out_size, void* d_ws, size_t ws_size,
                              hipStream_t stream) {
  (void)in_sizes; (void)n_in; (void)out_size; (void)ws_size;

  const float* in    = (const float*)d_in[0];
  const float* pars  = (const float*)d_in[1];
  const float* eWih0 = (const float*)d_in[2];
  const float* eWhh0 = (const float*)d_in[3];
  const float* eBih0 = (const float*)d_in[4];
  const float* eBhh0 = (const float*)d_in[5];
  const float* eWih1 = (const float*)d_in[6];
  const float* eWhh1 = (const float*)d_in[7];
  const float* eBih1 = (const float*)d_in[8];
  const float* eBhh1 = (const float*)d_in[9];
  const float* dWih0 = (const float*)d_in[10];
  const float* dWhh0 = (const float*)d_in[11];
  const float* dBih0 = (const float*)d_in[12];
  const float* dBhh0 = (const float*)d_in[13];
  const float* dWih1 = (const float*)d_in[14];
  const float* dWhh1 = (const float*)d_in[15];
  const float* dBih1 = (const float*)d_in[16];
  const float* dBhh1 = (const float*)d_in[17];
  const float* linW  = (const float*)d_in[18];
  const float* linB  = (const float*)d_in[19];
  const float* pdW1  = (const float*)d_in[20];
  const float* pdB1  = (const float*)d_in[21];
  const float* pdW2  = (const float*)d_in[22];
  const float* pdB2  = (const float*)d_in[23];
  float* out = (float*)d_out;

  // ---- workspace carve-out (~102 MB) ----
  char* wsp = (char*)d_ws;
  auto alloc = [&](size_t bytes) -> void* {
    void* p = (void*)wsp;
    wsp += (bytes + 255) & ~(size_t)255;
    return p;
  };
  __bf16* xin   = (__bf16*)alloc((size_t)SEQL * BATCH * I_SIZE * 2);  // bf16 input copy
  __bf16* wih0  = (__bf16*)alloc((size_t)G4H * I_SIZE * 2);
  __bf16* whh0  = (__bf16*)alloc((size_t)G4H * Hh * 2);
  __bf16* wih1  = (__bf16*)alloc((size_t)G4H * Hh * 2);
  __bf16* whh1  = (__bf16*)alloc((size_t)G4H * Hh * 2);
  __bf16* dwih0 = (__bf16*)alloc((size_t)G4H * DIN * 2);
  __bf16* dwhh0 = (__bf16*)alloc((size_t)G4H * Hh * 2);
  __bf16* dwih1 = (__bf16*)alloc((size_t)G4H * Hh * 2);
  __bf16* dwhh1 = (__bf16*)alloc((size_t)G4H * Hh * 2);
  __bf16* linw  = (__bf16*)alloc((size_t)I_SIZE * Hh * 2);
  __bf16* pdw2  = (__bf16*)alloc((size_t)HHs * HHs * 2);
  float* gates  = (float*)alloc((size_t)BATCH * G4H * 4);
  float* c0     = (float*)alloc((size_t)BATCH * Hh * 4);
  float* c1     = (float*)alloc((size_t)BATCH * Hh * 4);
  float* bias0  = (float*)alloc((size_t)G4H * 4);
  float* bias1  = (float*)alloc((size_t)G4H * 4);
  float* biasd0 = (float*)alloc((size_t)G4H * 4);
  float* biasd1 = (float*)alloc((size_t)G4H * 4);
  float* parft  = (float*)alloc((size_t)BATCH * HHs * 4);
  __bf16* h0    = (__bf16*)alloc((size_t)BATCH * Hh * 2);
  __bf16* h1    = (__bf16*)alloc((size_t)BATCH * Hh * 2);
  __bf16* zbf   = (__bf16*)alloc((size_t)BATCH * HHs * 2);
  __bf16* xdec  = (__bf16*)alloc((size_t)BATCH * DIN * 2);

  auto castw = [&](const float* s, __bf16* d, size_t n) {
    bf16_cast_kernel<<<(unsigned)((n + 255) / 256), 256, 0, stream>>>(s, d, (int)n);
  };
  auto gemm = [&](const __bf16* A0, int lda0, int K0, const __bf16* W0,
                  const __bf16* A1, int lda1, int K1, const __bf16* W1,
                  const float* bias, float* C, int ldc, int N) {
    dim3 grid(N / 256, BATCH / 32);
    gemm_wmma_kernel<<<grid, 128, 0, stream>>>(A0, lda0, K0, W0, A1, lda1, K1, W1,
                                               bias, C, ldc);
  };
  auto cell = [&](float* c, __bf16* h) {
    lstm_cell_kernel<<<(BATCH * Hh + 255) / 256, 256, 0, stream>>>(gates, c, h);
  };

  // ---- one-time conversions (all stay resident in 192MB L2) ----
  castw(in, xin, (size_t)SEQL * BATCH * I_SIZE);
  castw(eWih0, wih0, (size_t)G4H * I_SIZE);
  castw(eWhh0, whh0, (size_t)G4H * Hh);
  castw(eWih1, wih1, (size_t)G4H * Hh);
  castw(eWhh1, whh1, (size_t)G4H * Hh);
  castw(dWih0, dwih0, (size_t)G4H * DIN);
  castw(dWhh0, dwhh0, (size_t)G4H * Hh);
  castw(dWih1, dwih1, (size_t)G4H * Hh);
  castw(dWhh1, dwhh1, (size_t)G4H * Hh);
  castw(linW, linw, (size_t)I_SIZE * Hh);
  castw(pdW2, pdw2, (size_t)HHs * HHs);

  // ---- pre-summed biases b_ih + b_hh ----
  vec_add_kernel<<<(G4H + 255) / 256, 256, 0, stream>>>(eBih0, eBhh0, bias0, G4H);
  vec_add_kernel<<<(G4H + 255) / 256, 256, 0, stream>>>(eBih1, eBhh1, bias1, G4H);
  vec_add_kernel<<<(G4H + 255) / 256, 256, 0, stream>>>(dBih0, dBhh0, biasd0, G4H);
  vec_add_kernel<<<(G4H + 255) / 256, 256, 0, stream>>>(dBih1, dBhh1, biasd1, G4H);

  // ---- zero initial states ----
  fill_f32_kernel<<<(BATCH * Hh + 255) / 256, 256, 0, stream>>>(c0, BATCH * Hh);
  fill_f32_kernel<<<(BATCH * Hh + 255) / 256, 256, 0, stream>>>(c1, BATCH * Hh);
  fill_bf16_kernel<<<(BATCH * Hh + 255) / 256, 256, 0, stream>>>(h0, BATCH * Hh);
  fill_bf16_kernel<<<(BATCH * Hh + 255) / 256, 256, 0, stream>>>(h1, BATCH * Hh);

  // ---- par_dense MLP (constant across decode steps) ----
  pd1_kernel<<<(BATCH * HHs + 255) / 256, 256, 0, stream>>>(
      pars + (size_t)(SEQL - 1) * BATCH * Pp, pdW1, pdB1, zbf);
  gemm(zbf, HHs, HHs, pdw2, nullptr, 0, 0, nullptr, pdB2, parft, HHs, HHs);

  // ---- decoder input x0 = concat(input[-1], par_feat)  (bf16, ld = DIN) ----
  bf16_cast_strided_kernel<<<(BATCH * I_SIZE + 255) / 256, 256, 0, stream>>>(
      in + (size_t)(SEQL - 1) * BATCH * I_SIZE, I_SIZE, xdec, DIN, BATCH, I_SIZE);
  bf16_cast_strided_kernel<<<(BATCH * HHs + 255) / 256, 256, 0, stream>>>(
      parft, HHs, xdec + I_SIZE, DIN, BATCH, HHs);

  // ---- encoder: fused 2-layer recurrence over 256 steps ----
  for (int t = 0; t < SEQL; ++t) {
    gemm(xin + (size_t)t * BATCH * I_SIZE, I_SIZE, I_SIZE, wih0,
         h0, Hh, Hh, whh0, bias0, gates, G4H, G4H);
    cell(c0, h0);
    gemm(h0, Hh, Hh, wih1, h1, Hh, Hh, whh1, bias1, gates, G4H, G4H);
    cell(c1, h1);
  }

  // ---- autoregressive decoder, initial states = encoder finals ----
  for (int t = 0; t < TGTL; ++t) {
    gemm(xdec, DIN, DIN, dwih0, h0, Hh, Hh, dwhh0, biasd0, gates, G4H, G4H);
    cell(c0, h0);
    gemm(h0, Hh, Hh, dwih1, h1, Hh, Hh, dwhh1, biasd1, gates, G4H, G4H);
    cell(c1, h1);
    float* ot = out + (size_t)t * BATCH * I_SIZE;
    gemm(h1, Hh, Hh, linw, nullptr, 0, 0, nullptr, linB, ot, I_SIZE, I_SIZE);
    // feed output back as next-step input (cols 0..511; par_feat stays in 512..1023)
    bf16_cast_strided_kernel<<<(BATCH * I_SIZE + 255) / 256, 256, 0, stream>>>(
        ot, I_SIZE, xdec, DIN, BATCH, I_SIZE);
  }
}
